// EMAVectorQuantizer_85804856640196
// MI455X (gfx1250) — matrix operand order repaired and checked by reading, not compiled, not run on hardware
//
#include <hip/hip_runtime.h>
#include <hip/hip_bf16.h>

// ---------------------------------------------------------------------------
// EMA Vector Quantizer forward for MI455X (gfx1250, wave32, WMMA).
// Distances via split-precision bf16 WMMA: z.e ~= zh*eh + zh*el + zl*eh
// fused with an argmin epilogue; ||z||^2 dropped (constant per row).
// ---------------------------------------------------------------------------

typedef __attribute__((ext_vector_type(8)))  __bf16 bf16x8;
typedef __attribute__((ext_vector_type(16))) __bf16 bf16x16;
typedef __attribute__((ext_vector_type(8)))  float  v8f;

union V16 { bf16x16 v; bf16x8 h[2]; };

#define L_TOT 32768   // B*H*W = 32*32*32
#define D_DIM 256
#define N_E   2048
#define HWSZ  1024    // H*W

// ---------------------------------------------------------------------------
// Kernel 0a: transpose z [B,C,H,W] -> [L,D] and split into bf16 hi/lo planes.
// grid: 32768 blocks x 256 threads ; gid = d*L + l so reads are w-contiguous.
// ---------------------------------------------------------------------------
__global__ __launch_bounds__(256) void vq_prep_z(const float* __restrict__ z,
                                                 __bf16* __restrict__ zh,
                                                 __bf16* __restrict__ zl) {
  unsigned gid = blockIdx.x * 256u + threadIdx.x;   // [0, 8388608)
  unsigned d = gid >> 15;                           // [0,256)
  unsigned l = gid & (L_TOT - 1);                   // [0,32768)
  unsigned b = l >> 10;
  unsigned hw = l & (HWSZ - 1);
  float x = z[(b * D_DIM + d) * HWSZ + hw];
  __bf16 hi = (__bf16)x;
  __bf16 lo = (__bf16)(x - (float)hi);
  unsigned o = l * D_DIM + d;
  zh[o] = hi;
  zl[o] = lo;
}

// ---------------------------------------------------------------------------
// Kernel 0b: split embedding into bf16 hi/lo, compute ||e||^2 per row,
// zero the used-flags. grid: 2048 blocks x 256 threads.
// ---------------------------------------------------------------------------
__global__ __launch_bounds__(256) void vq_prep_emb(const float* __restrict__ emb,
                                                   __bf16* __restrict__ eh,
                                                   __bf16* __restrict__ el,
                                                   float* __restrict__ enorm,
                                                   float* __restrict__ used) {
  int n = blockIdx.x;
  int d = threadIdx.x;
  float x = emb[n * D_DIM + d];
  __bf16 hi = (__bf16)x;
  __bf16 lo = (__bf16)(x - (float)hi);
  eh[n * D_DIM + d] = hi;
  el[n * D_DIM + d] = lo;
  __shared__ float s[256];
  s[d] = x * x;
  __syncthreads();
  for (int k = 128; k > 0; k >>= 1) {
    if (d < k) s[d] += s[d + k];
    __syncthreads();
  }
  if (d == 0) {
    enorm[n] = s[0];
    used[n] = 0.0f;
  }
}

// ---------------------------------------------------------------------------
// Kernel 1: fused GEMM + argmin.  One wave per 16-row M-tile (2048 tiles,
// 8 waves/block -> 256 blocks).  A fragments (16 rows x full K=256, hi+lo)
// preloaded into VGPRs and reused across all 128 N-tiles.
// score[n] = ||e_n||^2 - 2 * <z_l, e_n>   (||z||^2 constant per row).
// ---------------------------------------------------------------------------
__global__ __launch_bounds__(256) void vq_argmin(const __bf16* __restrict__ zh,
                                                 const __bf16* __restrict__ zl,
                                                 const __bf16* __restrict__ eh,
                                                 const __bf16* __restrict__ el,
                                                 const float* __restrict__ enorm,
                                                 int*   __restrict__ idx_out,
                                                 float* __restrict__ idxf_out) {
  const int wave   = threadIdx.x >> 5;
  const int lane   = threadIdx.x & 31;
  const int tile_m = blockIdx.x * 8 + wave;          // [0, 2048)
  const int row    = tile_m * 16 + (lane & 15);      // A-matrix row (M)
  const int half   = lane >> 4;                      // 0: lanes 0-15, 1: 16-31

  // --- preload A fragments (16-bit A 16x32 layout: lane m holds row m,
  //     halves 0-7 = K kgrp..kgrp+7, halves 8-15 = K kgrp+16..kgrp+23,
  //     kgrp = 8*(lane>=16)) ---
  const __bf16* zrh = zh + (size_t)row * D_DIM;
  const __bf16* zrl = zl + (size_t)row * D_DIM;
  const int kgrp = half * 8;
  V16 a_hi[8], a_lo[8];
#pragma unroll
  for (int kc = 0; kc < 8; ++kc) {
    const int k0 = kc * 32 + kgrp;
    a_hi[kc].h[0] = *(const bf16x8*)(zrh + k0);
    a_hi[kc].h[1] = *(const bf16x8*)(zrh + k0 + 16);
    a_lo[kc].h[0] = *(const bf16x8*)(zrl + k0);
    a_lo[kc].h[1] = *(const bf16x8*)(zrl + k0 + 16);
  }

  float minval[8];
  int   minidx[8];
#pragma unroll
  for (int v = 0; v < 8; ++v) { minval[v] = 3.4e38f; minidx[v] = 0; }

  // --- scan all 128 N-tiles ---
  for (int nt = 0; nt < 128; ++nt) {
    const int n = nt * 16 + (lane & 15);             // B-matrix column (N)
    // B 32x16 bf16 layout: lane c holds column c&15; halves 0-7 = K kgrp..kgrp+7,
    // halves 8-15 = K kgrp+16..kgrp+23 (kgrp = 8*(lane>=16), same as A).
    const __bf16* erh = eh + (size_t)n * D_DIM + half * 8;
    const __bf16* erl = el + (size_t)n * D_DIM + half * 8;

    v8f acc = {};
#pragma unroll
    for (int kc = 0; kc < 8; ++kc) {
      V16 bh, bl;
      bh.h[0] = *(const bf16x8*)(erh + kc * 32);
      bh.h[1] = *(const bf16x8*)(erh + kc * 32 + 16);
      bl.h[0] = *(const bf16x8*)(erl + kc * 32);
      bl.h[1] = *(const bf16x8*)(erl + kc * 32 + 16);
      acc = __builtin_amdgcn_wmma_f32_16x16x32_bf16(false, a_hi[kc].v, false, bh.v,
                                                    (short)0, acc, false, false);
      acc = __builtin_amdgcn_wmma_f32_16x16x32_bf16(false, a_hi[kc].v, false, bl.v,
                                                    (short)0, acc, false, false);
      acc = __builtin_amdgcn_wmma_f32_16x16x32_bf16(false, a_lo[kc].v, false, bh.v,
                                                    (short)0, acc, false, false);
    }

    const float en = enorm[n];
#pragma unroll
    for (int v = 0; v < 8; ++v) {
      float sc = en - 2.0f * acc[v];
      if (sc < minval[v]) { minval[v] = sc; minidx[v] = n; }  // strict: keeps first
    }
  }

  // --- cross-lane argmin within each 16-lane half (xor 1,2,4,8 stays in half).
  // C-layout: VGPR v holds row M=v on lanes 0-15 and M=v+8 on lanes 16-31. ---
#pragma unroll
  for (int m = 1; m <= 8; m <<= 1) {
#pragma unroll
    for (int v = 0; v < 8; ++v) {
      float ov = __shfl_xor(minval[v], m, 32);
      int   oi = __shfl_xor(minidx[v], m, 32);
      if (ov < minval[v] || (ov == minval[v] && oi < minidx[v])) {
        minval[v] = ov; minidx[v] = oi;
      }
    }
  }

  if ((lane & 15) == 0) {
    const int base = tile_m * 16 + half * 8;
#pragma unroll
    for (int v = 0; v < 8; ++v) {
      idx_out[base + v]  = minidx[v];
      idxf_out[base + v] = (float)minidx[v];
    }
  }
}

// ---------------------------------------------------------------------------
// Kernel 2: gather e[idx] -> quantize (NCHW), deterministic per-row partial
// loss sums, used flags. grid: 32768 blocks x 256 threads.
// ---------------------------------------------------------------------------
__global__ __launch_bounds__(256) void vq_gather(const float* __restrict__ z,
                                                 const float* __restrict__ emb,
                                                 const int* __restrict__ idx,
                                                 float* __restrict__ quant,
                                                 float* __restrict__ partial,
                                                 float* __restrict__ used) {
  int l = blockIdx.x;
  int d = threadIdx.x;
  int b = l >> 10;
  int hw = l & (HWSZ - 1);
  int code = idx[l];
  float e  = emb[code * D_DIM + d];
  float zv = z[(b * D_DIM + d) * HWSZ + hw];
  quant[(b * D_DIM + d) * HWSZ + hw] = e;   // forward value of straight-through
  float diff = e - zv;
  __shared__ float s[256];
  s[d] = diff * diff;
  __syncthreads();
  for (int k = 128; k > 0; k >>= 1) {
    if (d < k) s[d] += s[d + k];
    __syncthreads();
  }
  if (d == 0) {
    partial[l] = s[0];
    used[code] = 1.0f;                      // benign race, all write 1.0
  }
}

// ---------------------------------------------------------------------------
// Kernel 3: deterministic finalize — sum 32768 loss partials and 2048 used
// flags in a fixed order. single block of 256 threads.
// ---------------------------------------------------------------------------
__global__ __launch_bounds__(256) void vq_finalize(const float* __restrict__ partial,
                                                   const float* __restrict__ used,
                                                   float* __restrict__ out_loss,
                                                   float* __restrict__ out_used) {
  int t = threadIdx.x;
  float ls = 0.0f, us = 0.0f;
  for (int i = t; i < L_TOT; i += 256) ls += partial[i];
  for (int i = t; i < N_E; i += 256)   us += used[i];
  __shared__ float sl[256], su[256];
  sl[t] = ls; su[t] = us;
  __syncthreads();
  for (int k = 128; k > 0; k >>= 1) {
    if (t < k) { sl[t] += sl[t + k]; su[t] += su[t + k]; }
    __syncthreads();
  }
  if (t == 0) {
    *out_loss = sl[0] / (float)((long)L_TOT * D_DIM);
    *out_used = su[0];
  }
}

// ---------------------------------------------------------------------------
// Launch
// ---------------------------------------------------------------------------
extern "C" void kernel_launch(void* const* d_in, const int* in_sizes, int n_in,
                              void* d_out, int out_size, void* d_ws, size_t ws_size,
                              hipStream_t stream) {
  const float* z   = (const float*)d_in[0];   // [32,256,32,32]
  const float* emb = (const float*)d_in[1];   // [2048,256]

  // workspace carve-up (all offsets 64B-aligned)
  char* ws = (char*)d_ws;
  __bf16* zh    = (__bf16*)(ws);                        // 16 MiB
  __bf16* zl    = (__bf16*)(ws + 16777216);             // 16 MiB
  __bf16* eh    = (__bf16*)(ws + 33554432);             // 1 MiB
  __bf16* el    = (__bf16*)(ws + 34603008);             // 1 MiB
  float*  enorm = (float*) (ws + 35651584);             // 8 KiB
  int*    idx   = (int*)   (ws + 35659776);             // 128 KiB
  float*  used  = (float*) (ws + 35790848);             // 8 KiB
  float*  part  = (float*) (ws + 35799040);             // 128 KiB

  // output layout: quantize | loss | index(float) | used
  float* out       = (float*)d_out;
  float* out_quant = out;                               // 8388608
  float* out_loss  = out + 8388608;                     // 1
  float* out_idxf  = out + 8388609;                     // 32768
  float* out_used  = out + 8388609 + 32768;             // 1

  vq_prep_z  <<<32768, 256, 0, stream>>>(z, zh, zl);
  vq_prep_emb<<<2048,  256, 0, stream>>>(emb, eh, el, enorm, used);
  vq_argmin  <<<256,   256, 0, stream>>>(zh, zl, eh, el, enorm, idx, out_idxf);
  vq_gather  <<<32768, 256, 0, stream>>>(z, emb, idx, out_quant, part, used);
  vq_finalize<<<1,     256, 0, stream>>>(part, used, out_loss, out_used);
}
